// AttendAndSpell_23072564314397
// MI455X (gfx1250) — compile-verified, matrix operand order checked by reading, not run on hardware
//
#include <hip/hip_runtime.h>

#define HID 512
#define VOCABN 1000
#define BATCH 64
#define RENC 1024
#define TSTEPS 200

typedef __attribute__((ext_vector_type(16))) __bf16 bf16x16;
typedef __attribute__((ext_vector_type(8)))  float  f32x8;
typedef __attribute__((ext_vector_type(4)))  unsigned int u32x4;
typedef __attribute__((ext_vector_type(4)))  float  f32x4;
typedef __attribute__((ext_vector_type(2)))  float  f32x2;

union Frag {
    bf16x16 v;
    u32x4 q[2];
    unsigned short u[16];
};

__device__ __forceinline__ float bf2f(unsigned short u) {
    return __uint_as_float(((unsigned int)u) << 16);
}
// hardware f32 -> bf16 (RNE) via __bf16 cast; backend emits v_cvt*bf16
__device__ __forceinline__ unsigned short f2bf(float f) {
    union { __bf16 b; unsigned short s; } u;
    u.b = (__bf16)f;
    return u.s;
}
// decode a packed pair of bf16 (one dword) into f32x2
__device__ __forceinline__ f32x2 bfpair(unsigned int w) {
    f32x2 r;
    r.x = __uint_as_float(w << 16);
    r.y = __uint_as_float(w & 0xFFFF0000u);
    return r;
}
__device__ __forceinline__ float sigm(float x) {
    return 1.0f / (1.0f + __expf(-x));
}

// Load a 16x32 bf16 A/B fragment from a row-major [rows x ld] bf16 matrix.
// ISA layout: lane<16 -> k-chunks {k0..k0+7, k0+16..k0+23}; lane>=16 -> +8.
__device__ __forceinline__ void load_frag(Frag& f, const unsigned short* base,
                                          int ld, int row, int k0, int lane) {
    const unsigned short* p = base + (size_t)row * ld + k0 + ((lane >> 4) << 3);
    f.q[0] = *reinterpret_cast<const u32x4*>(p);
    f.q[1] = *reinterpret_cast<const u32x4*>(p + 16);
}

// Same fragment, but source is f32 (converted inline to bf16 via hw cvt).
__device__ __forceinline__ void load_frag_f32(Frag& f, const float* base,
                                              int ld, int row, int k0, int lane) {
    const float* p = base + (size_t)row * ld + k0 + ((lane >> 4) << 3);
    f32x4 a = *reinterpret_cast<const f32x4*>(p);
    f32x4 b = *reinterpret_cast<const f32x4*>(p + 4);
    f32x4 c = *reinterpret_cast<const f32x4*>(p + 16);
    f32x4 d = *reinterpret_cast<const f32x4*>(p + 20);
#pragma unroll
    for (int i = 0; i < 4; ++i) {
        f.u[i]      = f2bf(a[i]);
        f.u[4 + i]  = f2bf(b[i]);
        f.u[8 + i]  = f2bf(c[i]);
        f.u[12 + i] = f2bf(d[i]);
    }
}

__device__ __forceinline__ f32x8 wmma_bf16(const Frag& a, const Frag& b, f32x8 c) {
    return __builtin_amdgcn_wmma_f32_16x16x32_bf16(
        false, a.v, false, b.v, (short)0, c, false, false);
}

// ---------------- conversion / init kernels ----------------

__global__ void cvt_f32_bf16_kernel(const float* __restrict__ src,
                                    unsigned short* __restrict__ dst, int n) {
    int i = blockIdx.x * 256 + threadIdx.x;
    if (i < n) dst[i] = f2bf(src[i]);
}

// w_ih0 is [2048][1512]; extract the context columns [1000..1511] as [2048][512] bf16
__global__ void cvt_wih0c_kernel(const float* __restrict__ w,
                                 unsigned short* __restrict__ dst) {
    int i = blockIdx.x * 256 + threadIdx.x;
    if (i < 2048 * 512) {
        int n = i >> 9, k = i & 511;
        dst[i] = f2bf(w[(size_t)n * (HID + VOCABN) + VOCABN + k]);
    }
}

__global__ void init_state_kernel(unsigned short* s0b, unsigned short* s1b,
                                  float* cs0, float* cs1) {
    int i = blockIdx.x * 256 + threadIdx.x;
    if (i < BATCH * HID) {
        s0b[i] = 0; s1b[i] = 0; cs0[i] = 0.f; cs1[i] = 0.f;
    }
}

// ---------------- hp = h @ psi_w^T + psi_b  (M=65536, N=512, K=512) ----------------
// Wave computes one 16-row M-tile x 4 N-tiles (64 cols). 4096 blocks x 8 waves.

__global__ __launch_bounds__(256) void hp_gemm_kernel(
    const float* __restrict__ h, const unsigned short* __restrict__ psiw,
    const float* __restrict__ psib, unsigned short* __restrict__ hp) {
    int lane = threadIdx.x & 31;
    int wave = (blockIdx.x << 3) + (threadIdx.x >> 5);
    int m0 = (wave >> 3) << 4;
    int n0 = (wave & 7) << 6;
    f32x8 z = {0.f, 0.f, 0.f, 0.f, 0.f, 0.f, 0.f, 0.f};
    f32x8 acc[4] = {z, z, z, z};
    int arow = m0 + (lane & 15);
    int brow = lane & 15;
    for (int k0 = 0; k0 < HID; k0 += 32) {
        Frag a; load_frag_f32(a, h, HID, arow, k0, lane);
#pragma unroll
        for (int g = 0; g < 4; ++g) {
            Frag b; load_frag(b, psiw, HID, n0 + (g << 4) + brow, k0, lane);
            acc[g] = wmma_bf16(a, b, acc[g]);
        }
    }
    int half = lane >> 4, nl = lane & 15;
#pragma unroll
    for (int g = 0; g < 4; ++g) {
        int n = n0 + (g << 4) + nl;
        float bias = psib[n];
#pragma unroll
        for (int j = 0; j < 8; ++j) {
            int row = m0 + j + (half << 3);
            hp[(size_t)row * HID + n] = f2bf(acc[g][j] + bias);
        }
    }
}

// ---------------- attention: one workgroup per batch element ----------------
// q = phi(s1_prev); e = hp_b . q; alpha = softmax(e); c = alpha . hp_b

__global__ __launch_bounds__(256) void attention_kernel(
    const unsigned short* __restrict__ s1b, const unsigned short* __restrict__ phiw,
    const float* __restrict__ phib, const unsigned short* __restrict__ hp,
    unsigned short* __restrict__ cb) {
    __shared__ float sh_s1[HID];
    __shared__ float sh_q[HID];
    __shared__ float sh_e[RENC];
    __shared__ float red[256];
    int b = blockIdx.x, tid = threadIdx.x;

    const unsigned short* s1p = s1b + (b << 9);
    for (int i = tid; i < HID; i += 256) sh_s1[i] = bf2f(s1p[i]);
    __syncthreads();

    // q = s1 @ phi_w^T + phi_b  (each thread: 2 output dims, packed f32x2 FMA)
#pragma unroll
    for (int jo = 0; jo < 2; ++jo) {
        int j = tid + (jo << 8);
        const unsigned short* wr = phiw + (size_t)j * HID;
        f32x2 acc2 = {phib[j], 0.f};
        for (int k = 0; k < HID; k += 8) {
            u32x4 w4 = *reinterpret_cast<const u32x4*>(wr + k);
#pragma unroll
            for (int e2 = 0; e2 < 4; ++e2) {
                f32x2 wv = bfpair(w4[e2]);
                f32x2 sv = {sh_s1[k + 2 * e2], sh_s1[k + 2 * e2 + 1]};
                acc2 += wv * sv;
            }
        }
        sh_q[j] = acc2.x + acc2.y;
    }
    __syncthreads();

    // e[r] = hp[b,r,:] . q   (each thread: 4 rows, packed f32x2 FMA)
    float emax = -1e30f;
#pragma unroll
    for (int ro = 0; ro < 4; ++ro) {
        int r = tid + (ro << 8);
        const unsigned short* hr = hp + ((size_t)(b * RENC + r) << 9);
        f32x2 acc2 = {0.f, 0.f};
        for (int k = 0; k < HID; k += 8) {
            u32x4 w4 = *reinterpret_cast<const u32x4*>(hr + k);
#pragma unroll
            for (int e2 = 0; e2 < 4; ++e2) {
                f32x2 hv = bfpair(w4[e2]);
                f32x2 qv = {sh_q[k + 2 * e2], sh_q[k + 2 * e2 + 1]};
                acc2 += hv * qv;
            }
        }
        float acc = acc2.x + acc2.y;
        sh_e[r] = acc;
        emax = fmaxf(emax, acc);
    }
    red[tid] = emax;
    __syncthreads();
    for (int s = 128; s > 0; s >>= 1) {
        if (tid < s) red[tid] = fmaxf(red[tid], red[tid + s]);
        __syncthreads();
    }
    float m = red[0];
    __syncthreads();

    float psum = 0.f;
#pragma unroll
    for (int ro = 0; ro < 4; ++ro) {
        int r = tid + (ro << 8);
        float p = __expf(sh_e[r] - m);
        sh_e[r] = p;
        psum += p;
    }
    red[tid] = psum;
    __syncthreads();
    for (int s = 128; s > 0; s >>= 1) {
        if (tid < s) red[tid] += red[tid + s];
        __syncthreads();
    }
    float inv = 1.0f / red[0];

    // c[d] = sum_r alpha[r] * hp[b,r,d]; thread owns dword pair d = {2*tid, 2*tid+1}
    // (consecutive threads read consecutive dwords -> fully coalesced rows)
    int d0 = tid << 1;
    f32x2 cacc = {0.f, 0.f};
    for (int r = 0; r < RENC; ++r) {
        float a = sh_e[r];
        const unsigned int* hr = reinterpret_cast<const unsigned int*>(
            hp + ((size_t)(b * RENC + r) << 9));
        f32x2 hv = bfpair(hr[tid]);
        f32x2 av = {a, a};
        cacc += av * hv;
    }
    cb[(b << 9) + d0]     = f2bf(cacc.x * inv);
    cb[(b << 9) + d0 + 1] = f2bf(cacc.y * inv);
}

// ---------------- LSTM cell: gates = A1@W1^T + A2@W2^T (+ gather) + biases ----------------
// 32 blocks x 128 threads = 128 waves: 4 M-tiles x 32 D-tiles.
// Each wave computes the i/f/g/o 16x16 tiles for its (batch-tile, dim-tile) patch
// and does the fused elementwise LSTM update.

__global__ __launch_bounds__(128) void lstm_cell_kernel(
    const unsigned short* __restrict__ A1, const unsigned short* __restrict__ W1,
    const unsigned short* __restrict__ A2, const unsigned short* __restrict__ W2,
    const float* __restrict__ b1, const float* __restrict__ b2,
    const float* __restrict__ gatherW, const int* __restrict__ y, int t,
    float* __restrict__ cs, unsigned short* __restrict__ hout,
    unsigned short* __restrict__ sseq) {
    int lane = threadIdx.x & 31;
    int wave = (blockIdx.x << 2) + (threadIdx.x >> 5);
    int m0 = (wave >> 5) << 4;
    int dd0 = (wave & 31) << 4;
    f32x8 z = {0.f, 0.f, 0.f, 0.f, 0.f, 0.f, 0.f, 0.f};
    f32x8 acc[4] = {z, z, z, z};
    int arow = m0 + (lane & 15);
    int brow = lane & 15;
    for (int k0 = 0; k0 < HID; k0 += 32) {
        Frag a; load_frag(a, A1, HID, arow, k0, lane);
#pragma unroll
        for (int g = 0; g < 4; ++g) {
            Frag b; load_frag(b, W1, HID, (g << 9) + dd0 + brow, k0, lane);
            acc[g] = wmma_bf16(a, b, acc[g]);
        }
    }
    for (int k0 = 0; k0 < HID; k0 += 32) {
        Frag a; load_frag(a, A2, HID, arow, k0, lane);
#pragma unroll
        for (int g = 0; g < 4; ++g) {
            Frag b; load_frag(b, W2, HID, (g << 9) + dd0 + brow, k0, lane);
            acc[g] = wmma_bf16(a, b, acc[g]);
        }
    }
    int half = lane >> 4, nl = lane & 15;
    int d = dd0 + nl;
    float bi = b1[d] + b2[d];
    float bf_ = b1[512 + d] + b2[512 + d];
    float bg = b1[1024 + d] + b2[1024 + d];
    float bo = b1[1536 + d] + b2[1536 + d];
#pragma unroll
    for (int j = 0; j < 8; ++j) {
        int bb = m0 + j + (half << 3);
        float gi = acc[0][j] + bi;
        float gf = acc[1][j] + bf_;
        float gg = acc[2][j] + bg;
        float go = acc[3][j] + bo;
        if (gatherW) {   // one-hot input contribution: column y[bb,t] of w_ih0
            int yv = y[bb * TSTEPS + t];
            const float* gw = gatherW + yv;
            gi += gw[(size_t)d * (HID + VOCABN)];
            gf += gw[(size_t)(512 + d) * (HID + VOCABN)];
            gg += gw[(size_t)(1024 + d) * (HID + VOCABN)];
            go += gw[(size_t)(1536 + d) * (HID + VOCABN)];
        }
        float iv = sigm(gi), fv = sigm(gf), gv = tanhf(gg), ov = sigm(go);
        int idx = (bb << 9) + d;
        float cn = fv * cs[idx] + iv * gv;
        cs[idx] = cn;
        float hn = ov * tanhf(cn);
        unsigned short hb = f2bf(hn);
        hout[idx] = hb;
        if (sseq) sseq[((size_t)(bb * TSTEPS + t) << 9) + d] = hb;
    }
}

// ---------------- out = s_seq @ out_w^T + out_b  (M=12800, N=1000, K=512) ----------------
// 6300 blocks x 8 waves = 50400 waves = 800 M-tiles x 63 N-tiles (last tile masked).

__global__ __launch_bounds__(256) void out_gemm_kernel(
    const unsigned short* __restrict__ sseq, const unsigned short* __restrict__ outw,
    const float* __restrict__ outb, float* __restrict__ out) {
    int lane = threadIdx.x & 31;
    int wave = (blockIdx.x << 3) + (threadIdx.x >> 5);
    int m0 = (wave / 63) << 4;
    int n0 = (wave % 63) << 4;
    f32x8 acc = {0.f, 0.f, 0.f, 0.f, 0.f, 0.f, 0.f, 0.f};
    int arow = m0 + (lane & 15);
    int nrow = n0 + (lane & 15);
    if (nrow > VOCABN - 1) nrow = VOCABN - 1;   // clamp; result masked at store
    for (int k0 = 0; k0 < HID; k0 += 32) {
        Frag a; load_frag(a, sseq, HID, arow, k0, lane);
        Frag b; load_frag(b, outw, HID, nrow, k0, lane);
        acc = wmma_bf16(a, b, acc);
    }
    int half = lane >> 4, nl = lane & 15;
    int n = n0 + nl;
    if (n < VOCABN) {
        float bias = outb[n];
#pragma unroll
        for (int j = 0; j < 8; ++j) {
            int row = m0 + j + (half << 3);
            out[(size_t)row * VOCABN + n] = acc[j] + bias;
        }
    }
}

// ---------------- host ----------------

extern "C" void kernel_launch(void* const* d_in, const int* in_sizes, int n_in,
                              void* d_out, int out_size, void* d_ws, size_t ws_size,
                              hipStream_t stream) {
    const float* h     = (const float*)d_in[0];
    const int*   y     = (const int*)d_in[1];
    const float* phi_w = (const float*)d_in[2];
    const float* phi_b = (const float*)d_in[3];
    const float* psi_w = (const float*)d_in[4];
    const float* psi_b = (const float*)d_in[5];
    const float* w_ih0 = (const float*)d_in[6];
    const float* w_hh0 = (const float*)d_in[7];
    const float* b_ih0 = (const float*)d_in[8];
    const float* b_hh0 = (const float*)d_in[9];
    const float* w_ih1 = (const float*)d_in[10];
    const float* w_hh1 = (const float*)d_in[11];
    const float* b_ih1 = (const float*)d_in[12];
    const float* b_hh1 = (const float*)d_in[13];
    const float* out_w = (const float*)d_in[14];
    const float* out_b = (const float*)d_in[15];
    float* out = (float*)d_out;

    char* ws = (char*)d_ws;
    size_t off = 0;
    auto alloc = [&](size_t bytes) -> void* {
        void* p = ws + off;
        off += (bytes + 255) & ~(size_t)255;
        return p;
    };
    unsigned short* hp      = (unsigned short*)alloc((size_t)BATCH * RENC * HID * 2); // 64 MB, L2-resident
    unsigned short* psi_bf  = (unsigned short*)alloc((size_t)HID * HID * 2);
    unsigned short* phi_bf  = (unsigned short*)alloc((size_t)HID * HID * 2);
    unsigned short* wih0c   = (unsigned short*)alloc((size_t)4 * HID * HID * 2);
    unsigned short* whh0_bf = (unsigned short*)alloc((size_t)4 * HID * HID * 2);
    unsigned short* wih1_bf = (unsigned short*)alloc((size_t)4 * HID * HID * 2);
    unsigned short* whh1_bf = (unsigned short*)alloc((size_t)4 * HID * HID * 2);
    unsigned short* outw_bf = (unsigned short*)alloc((size_t)VOCABN * HID * 2);
    unsigned short* s0b     = (unsigned short*)alloc((size_t)2 * BATCH * HID * 2); // ping-pong
    unsigned short* s1b     = (unsigned short*)alloc((size_t)2 * BATCH * HID * 2); // ping-pong
    unsigned short* cb      = (unsigned short*)alloc((size_t)BATCH * HID * 2);
    float*          cs0     = (float*)alloc((size_t)BATCH * HID * 4);
    float*          cs1     = (float*)alloc((size_t)BATCH * HID * 4);
    unsigned short* sseq    = (unsigned short*)alloc((size_t)BATCH * TSTEPS * HID * 2);
    (void)ws_size; (void)in_sizes; (void)n_in; (void)out_size;

    // Weight conversions to bf16 (one-time per call)
    cvt_f32_bf16_kernel<<<1024, 256, 0, stream>>>(psi_w, psi_bf, HID * HID);
    cvt_f32_bf16_kernel<<<1024, 256, 0, stream>>>(phi_w, phi_bf, HID * HID);
    cvt_f32_bf16_kernel<<<4096, 256, 0, stream>>>(w_hh0, whh0_bf, 4 * HID * HID);
    cvt_f32_bf16_kernel<<<4096, 256, 0, stream>>>(w_ih1, wih1_bf, 4 * HID * HID);
    cvt_f32_bf16_kernel<<<4096, 256, 0, stream>>>(w_hh1, whh1_bf, 4 * HID * HID);
    cvt_f32_bf16_kernel<<<2000, 256, 0, stream>>>(out_w, outw_bf, VOCABN * HID);
    cvt_wih0c_kernel<<<4096, 256, 0, stream>>>(w_ih0, wih0c);
    init_state_kernel<<<128, 256, 0, stream>>>(s0b, s1b, cs0, cs1);

    // hp = h @ psi_w^T + psi_b
    hp_gemm_kernel<<<4096, 256, 0, stream>>>(h, psi_bf, psi_b, hp);

    const int SB = BATCH * HID;  // elements per state buffer
    for (int t = 0; t < TSTEPS; ++t) {
        unsigned short* s0_prev = s0b + (t & 1) * SB;
        unsigned short* s0_new  = s0b + ((t + 1) & 1) * SB;
        unsigned short* s1_prev = s1b + (t & 1) * SB;
        unsigned short* s1_new  = s1b + ((t + 1) & 1) * SB;

        attention_kernel<<<BATCH, 256, 0, stream>>>(s1_prev, phi_bf, phi_b, hp, cb);

        lstm_cell_kernel<<<32, 128, 0, stream>>>(
            cb, wih0c, s0_prev, whh0_bf, b_ih0, b_hh0,
            w_ih0, y, t, cs0, s0_new, nullptr);

        lstm_cell_kernel<<<32, 128, 0, stream>>>(
            s0_new, wih1_bf, s1_prev, whh1_bf, b_ih1, b_hh1,
            nullptr, nullptr, t, cs1, s1_new, sseq);
    }

    // out = s_seq @ out_w^T + out_b
    out_gemm_kernel<<<6300, 256, 0, stream>>>(sseq, outw_bf, out_b, out);
}
